// gcn_UNet_30270929502678
// MI455X (gfx1250) — matrix operand order repaired and checked by reading, not compiled, not run on hardware
//
#include <hip/hip_runtime.h>
#include <hip/hip_bf16.h>

typedef __attribute__((ext_vector_type(16))) __bf16 v16bf;
typedef __attribute__((ext_vector_type(8)))  __bf16 v8bf;
typedef __attribute__((ext_vector_type(8)))  float  v8f;
typedef __attribute__((ext_vector_type(4)))  float  f32x4;

__device__ inline v16bf cvt16(f32x4 x0, f32x4 x1, f32x4 x2, f32x4 x3) {
  float t[16];
  *(f32x4*)(t + 0)  = x0;
  *(f32x4*)(t + 4)  = x1;
  *(f32x4*)(t + 8)  = x2;
  *(f32x4*)(t + 12) = x3;
  v16bf r;
#pragma unroll
  for (int e = 0; e < 16; ++e) r[e] = (__bf16)t[e];
  return r;
}

// ---------------------------------------------------------------------------
// Row degrees of A_hat = A + I :  d[i] = rsqrt(rowsum(A)+1+eps), A in f32
// (streaming read of the 268MB adjacency -> non-temporal)
// ---------------------------------------------------------------------------
__global__ __launch_bounds__(256) void deg_f32_k(const float* __restrict__ A,
                                                 int N, float* __restrict__ d) {
  __shared__ float red[256];
  const int row = blockIdx.x;
  const float* ar = A + (size_t)row * N;
  float s = 0.f;
  for (int c = threadIdx.x * 4; c < N; c += blockDim.x * 4) {
    f32x4 v = __builtin_nontemporal_load((const f32x4*)(ar + c));
    s += v.x + v.y + v.z + v.w;
  }
  red[threadIdx.x] = s;
  __syncthreads();
  for (int off = 128; off > 0; off >>= 1) {
    if (threadIdx.x < off) red[threadIdx.x] += red[threadIdx.x + off];
    __syncthreads();
  }
  if (threadIdx.x == 0) d[row] = rsqrtf(red[0] + 1.0f + 1e-10f);
}

__global__ __launch_bounds__(256) void deg_bf16_k(const __bf16* __restrict__ A,
                                                  int N, float* __restrict__ d) {
  __shared__ float red[256];
  const int row = blockIdx.x;
  const __bf16* ar = A + (size_t)row * N;
  float s = 0.f;
  for (int c = threadIdx.x * 8; c < N; c += blockDim.x * 8) {
    v8bf v = *(const v8bf*)(ar + c);
#pragma unroll
    for (int t = 0; t < 8; ++t) s += (float)v[t];
  }
  red[threadIdx.x] = s;
  __syncthreads();
  for (int off = 128; off > 0; off >>= 1) {
    if (threadIdx.x < off) red[threadIdx.x] += red[threadIdx.x + off];
    __syncthreads();
  }
  if (threadIdx.x == 0) d[row] = rsqrtf(red[0] + 1.0f + 1e-10f);
}

// ---------------------------------------------------------------------------
// Prep: ZT[n][i] = bf16( d[i] * dot(X[gather(i)], W[:,n]) )  (Z transposed)
// ---------------------------------------------------------------------------
__global__ __launch_bounds__(256) void prep_zt_k(const float* __restrict__ X, int Fin,
                                                 const float* __restrict__ W, int Fout,
                                                 const float* __restrict__ d,
                                                 const int* __restrict__ idx,
                                                 __bf16* __restrict__ ZT, long ldz, int M) {
  __shared__ float sw[1024];
  for (int t = threadIdx.x; t < Fin * Fout; t += blockDim.x) sw[t] = W[t];
  __syncthreads();
  const int i = blockIdx.x * blockDim.x + threadIdx.x;
  if (i >= M) return;
  const int src = idx ? idx[i] : i;
  float xr[32];
  for (int k = 0; k < Fin; ++k) xr[k] = X[(size_t)src * Fin + k];
  const float di = d[i];
  for (int n = 0; n < Fout; ++n) {
    float s = 0.f;
    for (int k = 0; k < Fin; ++k) s += xr[k] * sw[k * Fout + n];
    ZT[(size_t)n * ldz + i] = (__bf16)(di * s);
  }
}

// ---------------------------------------------------------------------------
// Main GEMM: Y += A[M x K] @ Z[K x 16*NNT]   (bf16 WMMA, split-K, atomics)
//   A_F32: A is binary-adjacency f32 -> NT loads + convert to bf16 (exact)
//   ZT is Z transposed: [16*NNT rows][K cols] bf16, row stride ldz
//   K-loop unrolled x2 (K=64/iter): all loads clustered ahead of the WMMAs.
//   NNT is compile-time so the hot loop is branch-free.
// ---------------------------------------------------------------------------
template <bool A_F32, int NNT>
__global__ __launch_bounds__(256) void gemm_wmma_k(const void* __restrict__ Aptr, long lda,
                                                   const __bf16* __restrict__ ZT, long ldz,
                                                   float* __restrict__ Y, int ldy,
                                                   int mtiles, int Kdim, int splitK) {
  const int lane = threadIdx.x & 31;
  const int wave = blockIdx.x * (blockDim.x >> 5) + (threadIdx.x >> 5);
  if (wave >= mtiles * splitK) return;          // wave-uniform: EXEC stays all-ones
  const int mt = wave % mtiles;
  const int ks = wave / mtiles;
  const int kper  = Kdim / splitK;              // multiple of 64
  const int k0beg = ks * kper;

  const bool hi   = lane >= 16;
  const int mrow  = mt * 16 + (lane & 15);      // A-frag row for this lane
  const int akoff = hi ? 16 : 0;                // A-frag K sub-offset
  const int bn    = lane & 15;                  // B-frag column
  const int bkoff = hi ? 16 : 0;                // B-frag K sub-offset

  v8f acc0 = {};
  v8f acc1 = {};
  const float*  arF = (const float*)Aptr + (size_t)mrow * lda + k0beg + akoff;
  const __bf16* arB = (const __bf16*)Aptr + (size_t)mrow * lda + k0beg + akoff;
  const __bf16* bp  = ZT + (size_t)bn * ldz + k0beg + bkoff;
  const __bf16* bq  = bp + (size_t)16 * ldz;    // second N-tile row block

  for (int it = 0; it < kper; it += 64) {
    v16bf a0, a1;
    // ---- cluster all loads for both K=32 chunks ----
    v16bf b00 = *(const v16bf*)(bp);
    v16bf b01 = *(const v16bf*)(bp + 32);
    if constexpr (A_F32) {
      f32x4 r0 = __builtin_nontemporal_load((const f32x4*)(arF + 0));
      f32x4 r1 = __builtin_nontemporal_load((const f32x4*)(arF + 4));
      f32x4 r2 = __builtin_nontemporal_load((const f32x4*)(arF + 8));
      f32x4 r3 = __builtin_nontemporal_load((const f32x4*)(arF + 12));
      f32x4 r4 = __builtin_nontemporal_load((const f32x4*)(arF + 32));
      f32x4 r5 = __builtin_nontemporal_load((const f32x4*)(arF + 36));
      f32x4 r6 = __builtin_nontemporal_load((const f32x4*)(arF + 40));
      f32x4 r7 = __builtin_nontemporal_load((const f32x4*)(arF + 44));
      a0 = cvt16(r0, r1, r2, r3);
      a1 = cvt16(r4, r5, r6, r7);
      arF += 64;
    } else {
      v8bf lo0 = *(const v8bf*)(arB);
      v8bf h80 = *(const v8bf*)(arB + 8);
      v8bf lo1 = *(const v8bf*)(arB + 32);
      v8bf h81 = *(const v8bf*)(arB + 40);
      a0 = __builtin_shufflevector(lo0, h80, 0, 1, 2, 3, 4, 5, 6, 7,
                                             8, 9, 10, 11, 12, 13, 14, 15);
      a1 = __builtin_shufflevector(lo1, h81, 0, 1, 2, 3, 4, 5, 6, 7,
                                             8, 9, 10, 11, 12, 13, 14, 15);
      arB += 64;
    }
    if constexpr (NNT > 1) {
      v16bf b10 = *(const v16bf*)(bq);
      v16bf b11 = *(const v16bf*)(bq + 32);
      acc0 = __builtin_amdgcn_wmma_f32_16x16x32_bf16(false, a0, false, b00,
                                                     (short)0, acc0, false, false);
      acc1 = __builtin_amdgcn_wmma_f32_16x16x32_bf16(false, a0, false, b10,
                                                     (short)0, acc1, false, false);
      acc0 = __builtin_amdgcn_wmma_f32_16x16x32_bf16(false, a1, false, b01,
                                                     (short)0, acc0, false, false);
      acc1 = __builtin_amdgcn_wmma_f32_16x16x32_bf16(false, a1, false, b11,
                                                     (short)0, acc1, false, false);
    } else {
      acc0 = __builtin_amdgcn_wmma_f32_16x16x32_bf16(false, a0, false, b00,
                                                     (short)0, acc0, false, false);
      acc0 = __builtin_amdgcn_wmma_f32_16x16x32_bf16(false, a1, false, b01,
                                                     (short)0, acc0, false, false);
    }
    bp += 64;
    bq += 64;
  }

  // C/D layout: lane l, vgpr v  ->  M = v + 8*(l>=16), N = l&15
  const int mbase = mt * 16 + (hi ? 8 : 0);
#pragma unroll
  for (int v = 0; v < 8; ++v)
    __hip_atomic_fetch_add(&Y[(size_t)(mbase + v) * ldy + bn], acc0[v],
                           __ATOMIC_RELAXED, __HIP_MEMORY_SCOPE_AGENT);
  if constexpr (NNT > 1) {
#pragma unroll
    for (int v = 0; v < 8; ++v)
      __hip_atomic_fetch_add(&Y[(size_t)(mbase + v) * ldy + 16 + bn], acc1[v],
                             __ATOMIC_RELAXED, __HIP_MEMORY_SCOPE_AGENT);
  }
}

// ---------------------------------------------------------------------------
// Epilogue: Xout[i][j] = relu( d[i] * (Y[i][j] + ZT[j][i]) )   (+Z = "+I" term)
// ---------------------------------------------------------------------------
__global__ __launch_bounds__(256) void epi_relu_k(const float* __restrict__ Y, int ldy,
                                                  const __bf16* __restrict__ ZT, long ldz,
                                                  const float* __restrict__ d,
                                                  float* __restrict__ Xo, int M, int F) {
  const int t = blockIdx.x * blockDim.x + threadIdx.x;
  if (t >= M * F) return;
  const int i = t / F, j = t % F;
  float v = d[i] * (Y[(size_t)i * ldy + j] + (float)ZT[(size_t)j * ldz + i]);
  Xo[(size_t)i * F + j] = v > 0.f ? v : 0.f;
}

__global__ __launch_bounds__(256) void softmax2_k(const float* __restrict__ Y, int ldy,
                                                  const __bf16* __restrict__ ZT, long ldz,
                                                  const float* __restrict__ d,
                                                  float* __restrict__ out, int M) {
  const int i = blockIdx.x * blockDim.x + threadIdx.x;
  if (i >= M) return;
  const float di = d[i];
  float v0 = di * (Y[(size_t)i * ldy + 0] + (float)ZT[i]);
  float v1 = di * (Y[(size_t)i * ldy + 1] + (float)ZT[ldz + i]);
  float m = fmaxf(v0, v1);
  float e0 = __expf(v0 - m), e1 = __expf(v1 - m);
  float inv = 1.f / (e0 + e1);
  out[2 * i]     = e0 * inv;
  out[2 * i + 1] = e1 * inv;
}

// ---------------------------------------------------------------------------
// scores[i] = dot(X[i,:], s)
// ---------------------------------------------------------------------------
__global__ __launch_bounds__(256) void scores_k(const float* __restrict__ X,
                                                const float* __restrict__ s,
                                                float* __restrict__ out, int M, int F) {
  const int i = blockIdx.x * blockDim.x + threadIdx.x;
  if (i >= M) return;
  float acc = 0.f;
  for (int k = 0; k < F; ++k) acc += X[(size_t)i * F + k] * s[k];
  out[i] = acc;
}

// ---------------------------------------------------------------------------
// Exact jax.lax.top_k: rank_i = #{j: s_j>s_i || (s_j==s_i && j<i)}; idx[rank]=i
// Scores live entirely in LDS (N<=8192 -> 32KB of the 320KB WGP pool).
// ---------------------------------------------------------------------------
__global__ __launch_bounds__(256) void topk_rank_k(const float* __restrict__ s, int N,
                                                   int K, int* __restrict__ idx) {
  __shared__ float sh[8192];
  for (int c = threadIdx.x; c < N; c += blockDim.x) sh[c] = s[c];
  __syncthreads();
  const int i = blockIdx.x * blockDim.x + threadIdx.x;
  if (i >= N) return;
  const float si = sh[i];
  int r = 0;
  for (int j = 0; j < N; ++j) {
    float sj = sh[j];
    r += (sj > si) || (sj == si && j < i);
  }
  if (r < K) idx[r] = i;
}

// ---------------------------------------------------------------------------
// A1p[r][c] = bf16( A[idx[r]][idx[c]] )   (0/1 entries: bf16 exact)
// ---------------------------------------------------------------------------
__global__ __launch_bounds__(256) void gather_a_k(const float* __restrict__ A, int N,
                                                  const int* __restrict__ idx, int K,
                                                  __bf16* __restrict__ Ap) {
  const int c = blockIdx.x * blockDim.x + threadIdx.x;
  const int r = blockIdx.y;
  if (c >= K) return;
  Ap[(size_t)r * K + c] = (__bf16)A[(size_t)idx[r] * N + idx[c]];
}

// srcByIdx=1 : dst[idx[p]] = src[idx[p]]  (unpool-2: mask rows of X2)
// srcByIdx=0 : dst[idx[p]] = src[p]       (unpool-1: scatter pooled rows)
__global__ __launch_bounds__(256) void scatter_rows_k(const float* __restrict__ src,
                                                      const int* __restrict__ idx,
                                                      float* __restrict__ dst,
                                                      int P, int F, int srcByIdx) {
  const int t = blockIdx.x * blockDim.x + threadIdx.x;
  if (t >= P * F) return;
  const int p = t / F, j = t % F;
  const int r = idx[p];
  const int sr = srcByIdx ? r : p;
  dst[(size_t)r * F + j] = src[(size_t)sr * F + j];
}

// ---------------------------------------------------------------------------
extern "C" void kernel_launch(void* const* d_in, const int* in_sizes, int n_in,
                              void* d_out, int out_size, void* d_ws, size_t ws_size,
                              hipStream_t stream) {
  const float* x  = (const float*)d_in[0];
  const float* a  = (const float*)d_in[1];
  const float* W1 = (const float*)d_in[2];
  const float* W2 = (const float*)d_in[3];
  const float* W3 = (const float*)d_in[4];
  const float* W4 = (const float*)d_in[5];
  const float* s1 = (const float*)d_in[6];
  const float* s2 = (const float*)d_in[7];

  const int FIN = 16;
  const int N   = in_sizes[0] / FIN;            // 8192
  const int F   = in_sizes[2] / FIN;            // 32
  const int C   = in_sizes[5] / F;              // 2
  const int K1  = (N + 1) / 2;                  // 4096
  const int K2  = (K1 + 1) / 2;                 // 2048
  const int SPLITK_N  = 8;                      // kper = 1024 (mult of 64)
  const int SPLITK_K1 = 16;                     // kper = 256  (mult of 64)

  // ---- carve workspace ----
  char* w = (char*)d_ws;
  size_t off = 0;
  auto carve = [&](size_t bytes) -> void* {
    void* p = w + off;
    off += (bytes + 255) & ~(size_t)255;
    return p;
  };
  __bf16* A1p = (__bf16*)carve((size_t)K1 * K1 * 2);
  size_t zeroBeg = off;
  float*  Y1  = (float*)carve((size_t)N * 32 * 4);
  float*  Y2  = (float*)carve((size_t)K1 * 32 * 4);
  float*  Y3  = (float*)carve((size_t)K1 * 32 * 4);
  float*  Y4  = (float*)carve((size_t)N * 16 * 4);
  float*  X3  = (float*)carve((size_t)K1 * F * 4);
  float*  X4  = (float*)carve((size_t)N * F * 4);
  __bf16* Z4T = (__bf16*)carve((size_t)16 * N * 2);
  size_t zeroEnd = off;
  float*  d1  = (float*)carve((size_t)N * 4);
  float*  d2  = (float*)carve((size_t)K1 * 4);
  __bf16* Z1T = (__bf16*)carve((size_t)32 * N * 2);
  __bf16* Z2T = (__bf16*)carve((size_t)32 * K1 * 2);
  __bf16* Z3T = (__bf16*)carve((size_t)32 * K1 * 2);
  float*  X1  = (float*)carve((size_t)N * F * 4);
  float*  X2  = (float*)carve((size_t)K1 * F * 4);
  float*  X3o = (float*)carve((size_t)K1 * F * 4);
  float*  sc  = (float*)carve((size_t)N * 4);
  int*    idx1 = (int*)carve((size_t)K1 * 4);
  int*    idx2 = (int*)carve((size_t)K2 * 4);

  hipMemsetAsync(w + zeroBeg, 0, zeroEnd - zeroBeg, stream);

  auto cdiv = [](int aa, int bb) { return (aa + bb - 1) / bb; };
  const int gemmBlocksN  = cdiv((N / 16) * SPLITK_N * 32, 256);    // layers 1, 4
  const int gemmBlocksK1 = cdiv((K1 / 16) * SPLITK_K1 * 32, 256);  // layers 2, 3

  // ---- Layer 1:  X1 = relu(A_norm @ (x @ W1)) ----
  deg_f32_k<<<N, 256, 0, stream>>>(a, N, d1);
  prep_zt_k<<<cdiv(N, 256), 256, 0, stream>>>(x, FIN, W1, F, d1, nullptr, Z1T, N, N);
  gemm_wmma_k<true, 2><<<gemmBlocksN, 256, 0, stream>>>(a, N, Z1T, N, Y1, 32,
                                                        N / 16, N, SPLITK_N);
  epi_relu_k<<<cdiv(N * F, 256), 256, 0, stream>>>(Y1, 32, Z1T, N, d1, X1, N, F);

  // ---- Pool 1 (top-K1 of X1@s1), gather A1p = a[idx1][:,idx1] as bf16 ----
  scores_k<<<cdiv(N, 256), 256, 0, stream>>>(X1, s1, sc, N, F);
  topk_rank_k<<<cdiv(N, 256), 256, 0, stream>>>(sc, N, K1, idx1);
  gather_a_k<<<dim3(cdiv(K1, 256), K1), 256, 0, stream>>>(a, N, idx1, K1, A1p);

  // ---- Layer 2:  X2 = relu(A1p_norm @ (X1[idx1] @ W2)) ----
  deg_bf16_k<<<K1, 256, 0, stream>>>(A1p, K1, d2);
  prep_zt_k<<<cdiv(K1, 256), 256, 0, stream>>>(X1, F, W2, F, d2, idx1, Z2T, K1, K1);
  gemm_wmma_k<false, 2><<<gemmBlocksK1, 256, 0, stream>>>(A1p, K1, Z2T, K1, Y2, 32,
                                                          K1 / 16, K1, SPLITK_K1);
  epi_relu_k<<<cdiv(K1 * F, 256), 256, 0, stream>>>(Y2, 32, Z2T, K1, d2, X2, K1, F);

  // ---- Pool 2 + unpool to K1 resolution (X3 = rows of X2 kept by idx2) ----
  scores_k<<<cdiv(K1, 256), 256, 0, stream>>>(X2, s2, sc, K1, F);
  topk_rank_k<<<cdiv(K1, 256), 256, 0, stream>>>(sc, K1, K2, idx2);
  scatter_rows_k<<<cdiv(K2 * F, 256), 256, 0, stream>>>(X2, idx2, X3, K2, F, 1);

  // ---- Layer 3:  X3o = relu(A1p_norm @ (X3 @ W3)) ----
  prep_zt_k<<<cdiv(K1, 256), 256, 0, stream>>>(X3, F, W3, F, d2, nullptr, Z3T, K1, K1);
  gemm_wmma_k<false, 2><<<gemmBlocksK1, 256, 0, stream>>>(A1p, K1, Z3T, K1, Y3, 32,
                                                          K1 / 16, K1, SPLITK_K1);
  epi_relu_k<<<cdiv(K1 * F, 256), 256, 0, stream>>>(Y3, 32, Z3T, K1, d2, X3o, K1, F);

  // ---- Unpool to N, Layer 4: softmax(A_norm @ (X4 @ W4)) ----
  scatter_rows_k<<<cdiv(K1 * F, 256), 256, 0, stream>>>(X3o, idx1, X4, K1, F, 0);
  prep_zt_k<<<cdiv(N, 256), 256, 0, stream>>>(X4, F, W4, C, d1, nullptr, Z4T, N, N);
  gemm_wmma_k<true, 1><<<gemmBlocksN, 256, 0, stream>>>(a, N, Z4T, N, Y4, 16,
                                                        N / 16, N, SPLITK_N);
  softmax2_k<<<cdiv(N, 256), 256, 0, stream>>>(Y4, 16, Z4T, N, d1, (float*)d_out, N);
}
